// Att_23502061043766
// MI455X (gfx1250) — matrix operand order repaired and checked by reading, not compiled, not run on hardware
//
#include <hip/hip_runtime.h>
#include <hip/hip_bf16.h>

#define NN_ 6144
#define DIMM 128

typedef __bf16 bf16;
typedef __attribute__((ext_vector_type(16))) __bf16 bf16x16;
typedef __attribute__((ext_vector_type(8)))  float  f32x8;
typedef __attribute__((ext_vector_type(4)))  unsigned int u32x4;
typedef __attribute__((ext_vector_type(8)))  int i32x8;
typedef __attribute__((ext_vector_type(4)))  int i32x4;

union FragAB { uint4 q[2]; bf16x16 v; };

__device__ __forceinline__ f32x8 wmma_bf16(bf16x16 a, bf16x16 b, f32x8 c) {
  return __builtin_amdgcn_wmma_f32_16x16x32_bf16(false, a, false, b, (short)0, c, false, false);
}

// ---- TDM: load a 2D tile [tile_rows=128][tile_k=64] of bf16 from a row-major
// [*, 6144] tensor into LDS, padding 16B after each 128B row (stride 72 elems).
__device__ __forceinline__ void tdm_load_tile(const bf16* gptr, unsigned lds_byte_off) {
  unsigned long long ga = (unsigned long long)(uintptr_t)gptr;
  u32x4 g0;
  g0[0] = 1u;                                   // count=1 valid descriptor
  g0[1] = lds_byte_off;                         // LDS dest (bytes)
  g0[2] = (unsigned)ga;                         // global addr [31:0]
  g0[3] = ((unsigned)(ga >> 32) & 0x01FFFFFFu)  // global addr [56:32]
          | 0x80000000u;                        // type=2 ("image")
  i32x8 g1;
  g1[0] = (int)((1u << 16)    // data_size = 2 bytes
              | (1u << 20)    // pad_enable
              | (4u << 22)    // pad_interval: 32 DWORDs (=128B row)
              | (3u << 25));  // pad_amount: 4 DWORDs (=16B)
  g1[1] = (int)(((unsigned)NN_ & 0xFFFFu) << 16);            // tensor_dim0[15:0]
  g1[2] = (int)((((unsigned)NN_ >> 16) & 0xFFFFu)            // tensor_dim0[31:16]
              | (((unsigned)NN_ & 0xFFFFu) << 16));          // tensor_dim1[15:0]
  g1[3] = (int)((((unsigned)NN_ >> 16) & 0xFFFFu)            // tensor_dim1[31:16]
              | (64u << 16));                                // tile_dim0 = 64
  g1[4] = (int)128u;                                         // tile_dim1 = 128, tile_dim2 = 0
  g1[5] = (int)(unsigned)NN_;                                // tensor_dim0_stride[31:0]
  g1[6] = 0;                                                 // stride hi / dim1_stride lo
  g1[7] = 0;
  i32x4 z4 = {0, 0, 0, 0};
  i32x8 z8 = {0, 0, 0, 0, 0, 0, 0, 0};
  __builtin_amdgcn_tensor_load_to_lds(g0, g1, z4, z4, z8, 0);
}

__device__ __forceinline__ unsigned lds_off(const bf16* p) {
  return (unsigned)(uintptr_t)p;   // LDS aperture: byte offset lives in addr[31:0]
}

// ---------------- K0: f32 -> bf16 convert ----------------
__global__ void convert_kernel(const float* __restrict__ src, bf16* __restrict__ dst, long long n) {
  long long i = ((long long)blockIdx.x * blockDim.x + threadIdx.x) * 4;
  if (i + 4 <= n) {
    float4 f = *(const float4*)(src + i);
    union { bf16 b[4]; unsigned long long u; } p;
    p.b[0] = (bf16)f.x; p.b[1] = (bf16)f.y; p.b[2] = (bf16)f.z; p.b[3] = (bf16)f.w;
    *(unsigned long long*)(dst + i) = p.u;
  } else {
    for (long long j = i; j < n; ++j) dst[j] = (bf16)src[j];
  }
}

// ---------------- K1: q/k/v projection GEMM (WMMA) ----------------
__global__ __launch_bounds__(256) void proj_kernel(
    const bf16* __restrict__ hbf,
    const bf16* __restrict__ w0, const bf16* __restrict__ w1, const bf16* __restrict__ w2,
    const float* __restrict__ b0, const float* __restrict__ b1, const float* __restrict__ b2,
    bf16* __restrict__ qo, bf16* __restrict__ ko, bf16* __restrict__ vto) {
  constexpr int LDT = DIMM + 8;
  __shared__ __align__(16) bf16 As[128 * LDT];
  __shared__ __align__(16) bf16 Bs[128 * LDT];
  const int m = blockIdx.y;
  const bf16* W = (m == 0) ? w0 : (m == 1) ? w1 : w2;
  const float* B = (m == 0) ? b0 : (m == 1) ? b1 : b2;
  int tid = threadIdx.x, wid = tid >> 5, lane = tid & 31;
  int n16 = lane & 15, kh = lane >> 4;
  int waveM = wid & 3, waveN = wid >> 2;
  int gM0 = blockIdx.x * 128;
  for (int i = tid; i < 128 * 16; i += 256) {
    int row = i >> 4, c = (i & 15) * 8;
    *(uint4*)(&As[row * LDT + c]) = *(const uint4*)(&hbf[(size_t)(gM0 + row) * DIMM + c]);
    *(uint4*)(&Bs[row * LDT + c]) = *(const uint4*)(&W[(size_t)row * DIMM + c]);
  }
  __syncthreads();
  f32x8 acc[2][4] = {};
  for (int ks = 0; ks < DIMM; ks += 32) {
    FragAB a[2], b[4];
    for (int mt = 0; mt < 2; ++mt) {
      int row = waveM * 32 + mt * 16 + n16;
      a[mt].q[0] = *(const uint4*)(&As[row * LDT + ks + kh * 8]);
      a[mt].q[1] = *(const uint4*)(&As[row * LDT + ks + 16 + kh * 8]);
    }
    for (int nt = 0; nt < 4; ++nt) {
      int j = waveN * 64 + nt * 16 + n16;
      b[nt].q[0] = *(const uint4*)(&Bs[j * LDT + ks + kh * 8]);
      b[nt].q[1] = *(const uint4*)(&Bs[j * LDT + ks + 16 + kh * 8]);
    }
    for (int mt = 0; mt < 2; ++mt)
      for (int nt = 0; nt < 4; ++nt)
        acc[mt][nt] = wmma_bf16(a[mt].v, b[nt].v, acc[mt][nt]);
  }
  const float qscale = 0.08838834764831845f; // 1/sqrt(128)
  for (int mt = 0; mt < 2; ++mt)
    for (int nt = 0; nt < 4; ++nt) {
      int col = waveN * 64 + nt * 16 + n16;
      float bias = B[col];
      for (int r = 0; r < 8; ++r) {
        int row = gM0 + waveM * 32 + mt * 16 + kh * 8 + r;
        float v = acc[mt][nt][r] + bias;
        if (m == 0)      qo[(size_t)row * DIMM + col] = (bf16)(v * qscale);
        else if (m == 1) ko[(size_t)row * DIMM + col] = (bf16)v;
        else             vto[(size_t)col * NN_ + row] = (bf16)v;
      }
    }
}

// ---------------- K2: fused dual GEMM via TDM double buffering ----------------
// pe = e@Pw.T + Pb ; gate = sigmoid(e@Gw.T + Gb)
__global__ __launch_bounds__(256) void dual_gemm_kernel(
    const bf16* __restrict__ Ebf, const bf16* __restrict__ Pwb, const bf16* __restrict__ Gwb,
    const float* __restrict__ Pb, const float* __restrict__ Gb,
    float* __restrict__ pe, bf16* __restrict__ gate) {
  constexpr int KB = 64, LDT = KB + 8;                  // 72 elems = 144B row stride
  __shared__ __align__(16) bf16 As[2][128 * LDT];
  __shared__ __align__(16) bf16 BPs[2][128 * LDT];
  __shared__ __align__(16) bf16 BGs[2][128 * LDT];
  int tid = threadIdx.x, wid = tid >> 5, lane = tid & 31;
  int n16 = lane & 15, kh = lane >> 4;
  int waveM = wid & 3, waveN = wid >> 2;                // 4 x 2, wave tile 32x64
  int gM0 = blockIdx.x * 128, gN0 = blockIdx.y * 128;
  const size_t aBase = (size_t)gM0 * NN_;
  const size_t bBase = (size_t)gN0 * NN_;
  f32x8 accP[2][4] = {}, accG[2][4] = {};

  if (wid == 0) {                       // wave 0 drives the Tensor Data Mover
    tdm_load_tile(&Ebf[aBase], lds_off(&As[0][0]));
    tdm_load_tile(&Pwb[bBase], lds_off(&BPs[0][0]));
    tdm_load_tile(&Gwb[bBase], lds_off(&BGs[0][0]));
    __builtin_amdgcn_s_wait_tensorcnt(0);
  }
  __syncthreads();

  constexpr int NIT = NN_ / KB;         // 96
  for (int it = 0; it < NIT; ++it) {
    int buf = it & 1;
    if (wid == 0 && it + 1 < NIT) {     // prefetch next K-slab into other buffer
      size_t k1 = (size_t)(it + 1) * KB;
      tdm_load_tile(&Ebf[aBase + k1], lds_off(&As[buf ^ 1][0]));
      tdm_load_tile(&Pwb[bBase + k1], lds_off(&BPs[buf ^ 1][0]));
      tdm_load_tile(&Gwb[bBase + k1], lds_off(&BGs[buf ^ 1][0]));
    }
    for (int ks = 0; ks < KB; ks += 32) {
      FragAB a[2], bp[4], bg[4];
      for (int mt = 0; mt < 2; ++mt) {
        int row = waveM * 32 + mt * 16 + n16;
        a[mt].q[0] = *(const uint4*)(&As[buf][row * LDT + ks + kh * 8]);
        a[mt].q[1] = *(const uint4*)(&As[buf][row * LDT + ks + 16 + kh * 8]);
      }
      for (int nt = 0; nt < 4; ++nt) {
        int j = waveN * 64 + nt * 16 + n16;
        bp[nt].q[0] = *(const uint4*)(&BPs[buf][j * LDT + ks + kh * 8]);
        bp[nt].q[1] = *(const uint4*)(&BPs[buf][j * LDT + ks + 16 + kh * 8]);
        bg[nt].q[0] = *(const uint4*)(&BGs[buf][j * LDT + ks + kh * 8]);
        bg[nt].q[1] = *(const uint4*)(&BGs[buf][j * LDT + ks + 16 + kh * 8]);
      }
      for (int mt = 0; mt < 2; ++mt)
        for (int nt = 0; nt < 4; ++nt) {
          accP[mt][nt] = wmma_bf16(a[mt].v, bp[nt].v, accP[mt][nt]);
          accG[mt][nt] = wmma_bf16(a[mt].v, bg[nt].v, accG[mt][nt]);
        }
    }
    if (wid == 0) __builtin_amdgcn_s_wait_tensorcnt(0);
    __syncthreads();
  }

  for (int mt = 0; mt < 2; ++mt)
    for (int nt = 0; nt < 4; ++nt) {
      int col = gN0 + waveN * 64 + nt * 16 + n16;
      float pb = Pb[col], gb = Gb[col];
      for (int r = 0; r < 8; ++r) {
        int row = gM0 + waveM * 32 + mt * 16 + kh * 8 + r;
        size_t idx = (size_t)row * NN_ + col;
        pe[idx] = accP[mt][nt][r] + pb;
        float g = accG[mt][nt][r] + gb;
        gate[idx] = (bf16)(1.0f / (1.0f + __expf(-g)));
      }
    }
}

// ---------------- K3: S = q@k.T + pe (in place on pe buffer) ----------------
__global__ __launch_bounds__(256) void scores_kernel(
    const bf16* __restrict__ qb, const bf16* __restrict__ kb, float* __restrict__ S) {
  constexpr int LDT = DIMM + 8;
  __shared__ __align__(16) bf16 Qs[128 * LDT];
  __shared__ __align__(16) bf16 Ks[128 * LDT];
  int tid = threadIdx.x, wid = tid >> 5, lane = tid & 31;
  int n16 = lane & 15, kh = lane >> 4;
  int waveM = wid & 3, waveN = wid >> 2;
  int gM0 = blockIdx.x * 128, gN0 = blockIdx.y * 128;
  for (int i = tid; i < 128 * 16; i += 256) {
    int row = i >> 4, c = (i & 15) * 8;
    *(uint4*)(&Qs[row * LDT + c]) = *(const uint4*)(&qb[(size_t)(gM0 + row) * DIMM + c]);
    *(uint4*)(&Ks[row * LDT + c]) = *(const uint4*)(&kb[(size_t)(gN0 + row) * DIMM + c]);
  }
  __syncthreads();
  f32x8 acc[2][4] = {};
  for (int ks = 0; ks < DIMM; ks += 32) {
    FragAB a[2], b[4];
    for (int mt = 0; mt < 2; ++mt) {
      int row = waveM * 32 + mt * 16 + n16;
      a[mt].q[0] = *(const uint4*)(&Qs[row * LDT + ks + kh * 8]);
      a[mt].q[1] = *(const uint4*)(&Qs[row * LDT + ks + 16 + kh * 8]);
    }
    for (int nt = 0; nt < 4; ++nt) {
      int j = waveN * 64 + nt * 16 + n16;
      b[nt].q[0] = *(const uint4*)(&Ks[j * LDT + ks + kh * 8]);
      b[nt].q[1] = *(const uint4*)(&Ks[j * LDT + ks + 16 + kh * 8]);
    }
    for (int mt = 0; mt < 2; ++mt)
      for (int nt = 0; nt < 4; ++nt)
        acc[mt][nt] = wmma_bf16(a[mt].v, b[nt].v, acc[mt][nt]);
  }
  for (int mt = 0; mt < 2; ++mt)
    for (int nt = 0; nt < 4; ++nt) {
      int col = gN0 + waveN * 64 + nt * 16 + n16;
      for (int r = 0; r < 8; ++r) {
        int row = gM0 + waveM * 32 + mt * 16 + kh * 8 + r;
        size_t idx = (size_t)row * NN_ + col;
        S[idx] = S[idx] + acc[mt][nt][r];
      }
    }
}

// ---------------- K4: softmax + JAX threefry dropout + gate ----------------
__device__ __forceinline__ unsigned rotl32(unsigned x, int r) { return (x << r) | (x >> (32 - r)); }

__device__ __forceinline__ bool threefry_keep(size_t idx) {
  const unsigned half = (unsigned)(((size_t)NN_ * NN_) / 2);
  unsigned x0, x1; bool first;
  if (idx < (size_t)half) { x0 = (unsigned)idx; x1 = (unsigned)idx + half; first = true; }
  else { x0 = (unsigned)idx - half; x1 = (unsigned)idx; first = false; }
  const unsigned ks0 = 0u, ks1 = 42u, ks2 = 0x1BD11BDAu ^ ks0 ^ ks1;
  x0 += ks0; x1 += ks1;
  #define TF4(r0,r1,r2,r3) \
    x0 += x1; x1 = rotl32(x1, r0); x1 ^= x0; \
    x0 += x1; x1 = rotl32(x1, r1); x1 ^= x0; \
    x0 += x1; x1 = rotl32(x1, r2); x1 ^= x0; \
    x0 += x1; x1 = rotl32(x1, r3); x1 ^= x0;
  TF4(13,15,26,6)  x0 += ks1; x1 += ks2 + 1u;
  TF4(17,29,16,24) x0 += ks2; x1 += ks0 + 2u;
  TF4(13,15,26,6)  x0 += ks0; x1 += ks1 + 3u;
  TF4(17,29,16,24) x0 += ks1; x1 += ks2 + 4u;
  TF4(13,15,26,6)  x0 += ks2; x1 += ks0 + 5u;
  #undef TF4
  unsigned bits = first ? x0 : x1;
  float u = __uint_as_float((bits >> 9) | 0x3f800000u) - 1.0f;
  return u < 0.8f;
}

__global__ __launch_bounds__(256) void softmax_kernel(
    const float* __restrict__ S, const bf16* __restrict__ gate, bf16* __restrict__ P) {
  __shared__ float rowv[NN_];
  __shared__ float red[256];
  int row = blockIdx.x, tid = threadIdx.x;
  size_t base = (size_t)row * NN_;
  float lmax = -3.4e38f;
  for (int j = tid; j < NN_; j += 256) { float s = S[base + j]; rowv[j] = s; lmax = fmaxf(lmax, s); }
  red[tid] = lmax; __syncthreads();
  for (int s = 128; s > 0; s >>= 1) { if (tid < s) red[tid] = fmaxf(red[tid], red[tid + s]); __syncthreads(); }
  float m = red[0]; __syncthreads();
  float lsum = 0.f;
  for (int j = tid; j < NN_; j += 256) { float ev = __expf(rowv[j] - m); rowv[j] = ev; lsum += ev; }
  red[tid] = lsum; __syncthreads();
  for (int s = 128; s > 0; s >>= 1) { if (tid < s) red[tid] += red[tid + s]; __syncthreads(); }
  float inv = 1.0f / red[0];
  for (int j = tid; j < NN_; j += 256) {
    size_t idx = base + j;
    float a = threefry_keep(idx) ? rowv[j] * inv * 1.25f : 0.0f;
    P[idx] = (bf16)(a * (float)gate[idx]);
  }
}

// ---------------- K5: out = P @ v  (v stored transposed, bf16) ----------------
__global__ __launch_bounds__(256) void out_kernel(
    const bf16* __restrict__ P, const bf16* __restrict__ vt, float* __restrict__ out) {
  constexpr int KB = 64, LDT = KB + 8;
  __shared__ __align__(16) bf16 As[64 * LDT];
  __shared__ __align__(16) bf16 Bs[128 * LDT];
  int tid = threadIdx.x, wid = tid >> 5, lane = tid & 31;
  int n16 = lane & 15, kh = lane >> 4;
  int waveM = wid & 1, waveN = wid >> 1;
  int gM0 = blockIdx.x * 64;
  f32x8 acc[2][2] = {};
  for (int k0 = 0; k0 < NN_; k0 += KB) {
    __syncthreads();
    int r = tid >> 3, c = (tid & 7) * 8;
    for (int p = 0; p < 2; ++p)
      *(uint4*)(&As[(r + p * 32) * LDT + c]) = *(const uint4*)(&P[(size_t)(gM0 + r + p * 32) * NN_ + k0 + c]);
    for (int p = 0; p < 4; ++p)
      *(uint4*)(&Bs[(r + p * 32) * LDT + c]) = *(const uint4*)(&vt[(size_t)(r + p * 32) * NN_ + k0 + c]);
    __syncthreads();
    for (int ks = 0; ks < KB; ks += 32) {
      FragAB a[2], b[2];
      for (int mt = 0; mt < 2; ++mt) {
        int row = waveM * 32 + mt * 16 + n16;
        a[mt].q[0] = *(const uint4*)(&As[row * LDT + ks + kh * 8]);
        a[mt].q[1] = *(const uint4*)(&As[row * LDT + ks + 16 + kh * 8]);
      }
      for (int nt = 0; nt < 2; ++nt) {
        int d = waveN * 32 + nt * 16 + n16;
        b[nt].q[0] = *(const uint4*)(&Bs[d * LDT + ks + kh * 8]);
        b[nt].q[1] = *(const uint4*)(&Bs[d * LDT + ks + 16 + kh * 8]);
      }
      for (int mt = 0; mt < 2; ++mt)
        for (int nt = 0; nt < 2; ++nt)
          acc[mt][nt] = wmma_bf16(a[mt].v, b[nt].v, acc[mt][nt]);
    }
  }
  for (int mt = 0; mt < 2; ++mt)
    for (int nt = 0; nt < 2; ++nt) {
      int col = waveN * 32 + nt * 16 + n16;
      for (int r = 0; r < 8; ++r) {
        int row = gM0 + waveM * 32 + mt * 16 + kh * 8 + r;
        out[(size_t)row * DIMM + col] = acc[mt][nt][r];
      }
    }
}

// ---------------- launch ----------------
extern "C" void kernel_launch(void* const* d_in, const int* in_sizes, int n_in,
                              void* d_out, int out_size, void* d_ws, size_t ws_size,
                              hipStream_t stream) {
  const float* h  = (const float*)d_in[0];
  const float* e  = (const float*)d_in[1];
  const float* Qw = (const float*)d_in[2];
  const float* Qb = (const float*)d_in[3];
  const float* Kw = (const float*)d_in[4];
  const float* Kb = (const float*)d_in[5];
  const float* Vw = (const float*)d_in[6];
  const float* Vb = (const float*)d_in[7];
  const float* Pw = (const float*)d_in[8];
  const float* Pb = (const float*)d_in[9];
  const float* Gw = (const float*)d_in[10];
  const float* Gb = (const float*)d_in[11];
  float* out = (float*)d_out;

  const size_t NN = (size_t)NN_ * NN_;
  bf16* ebf  = (bf16*)d_ws;
  bf16* pwb  = ebf + NN;
  bf16* gwb  = pwb + NN;
  float* pe  = (float*)(gwb + NN);
  bf16* gate = (bf16*)(pe + NN);
  bf16* hbf  = gate + NN;
  bf16* qwb  = hbf + (size_t)NN_ * DIMM;
  bf16* kwb  = qwb + DIMM * DIMM;
  bf16* vwb  = kwb + DIMM * DIMM;
  bf16* qbm  = vwb + DIMM * DIMM;
  bf16* kbm  = qbm + (size_t)NN_ * DIMM;
  bf16* vtm  = kbm + (size_t)NN_ * DIMM;
  bf16* Pmat = ebf; // reuse: e_bf dead after dual_gemm

  auto cvt = [&](const float* s, bf16* d, long long n) {
    int blocks = (int)((n / 4 + 255) / 256);
    convert_kernel<<<blocks, 256, 0, stream>>>(s, d, n);
  };
  cvt(e, ebf, (long long)NN);
  cvt(Pw, pwb, (long long)NN);
  cvt(Gw, gwb, (long long)NN);
  cvt(h, hbf, (long long)NN_ * DIMM);
  cvt(Qw, qwb, DIMM * DIMM);
  cvt(Kw, kwb, DIMM * DIMM);
  cvt(Vw, vwb, DIMM * DIMM);

  proj_kernel<<<dim3(NN_ / 128, 3), 256, 0, stream>>>(hbf, qwb, kwb, vwb, Qb, Kb, Vb, qbm, kbm, vtm);
  dual_gemm_kernel<<<dim3(NN_ / 128, NN_ / 128), 256, 0, stream>>>(ebf, pwb, gwb, Pb, Gb, pe, gate);
  scores_kernel<<<dim3(NN_ / 128, NN_ / 128), 256, 0, stream>>>(qbm, kbm, pe);
  softmax_kernel<<<NN_, 256, 0, stream>>>(pe, gate, Pmat);
  out_kernel<<<NN_ / 64, 256, 0, stream>>>(Pmat, vtm, out);
}
